// PingHead_12790412607633
// MI455X (gfx1250) — compile-verified, matrix operand order checked by reading, not run on hardware
//
#include <hip/hip_runtime.h>
#include <hip/hip_bf16.h>

typedef __bf16 bf16;
typedef __attribute__((ext_vector_type(16))) __bf16 v16bf;
typedef __attribute__((ext_vector_type(8)))  __bf16 v8bf;
typedef __attribute__((ext_vector_type(4)))  __bf16 v4bf;
typedef __attribute__((ext_vector_type(8)))  float  v8f;
typedef __attribute__((ext_vector_type(4)))  float  v4f;

#define H_DIM  1024
#define G_DIM  128
#define N_DIM  384      // 3 * G_DIM
#define TILE_M 32       // rows of x per workgroup
#define KSTEP  32       // K per WMMA (bf16 16x16x32)
#define KCHUNK 128      // K staged in LDS per barrier pair (4 WMMA sub-steps)

// ---------------------------------------------------------------------------
// Kernel 1: convert weight_ih (384 x 1024, f32) to bf16 in d_ws.
// ---------------------------------------------------------------------------
__global__ void wih_to_bf16(const float* __restrict__ w, bf16* __restrict__ o, int n) {
    int i = blockIdx.x * blockDim.x + threadIdx.x;
    if (i < n) o[i] = (bf16)w[i];
}

// Build a v16bf A/B fragment from two contiguous 16-byte runs
// (ISA 7.12.2 16-bit layout: K in [kb,kb+8) and [kb+16,kb+24)).
static __device__ inline v16bf load_frag(const bf16* p) {
    v8bf lo = *(const v8bf*)(p);
    v8bf hi = *(const v8bf*)(p + 16);
    return __builtin_shufflevector(lo, hi,
            0,1,2,3,4,5,6,7,8,9,10,11,12,13,14,15);
}

// ---------------------------------------------------------------------------
// Kernel 2: fused GEMM (bf16 WMMA) + GRU-gate epilogue + lin_w reduction.
// Grid: M/32 blocks of 256 threads. 8 waves each own ALL 32 rows (two A
// fragments) x 3 N-tiles, so every global B fragment feeds two WMMAs.
// x is staged in 128-K chunks (one barrier pair per 24 WMMAs) with
// non-temporal loads so the streamed 268 MB of x does not evict the
// L2-resident bf16 weight that every workgroup re-reads.
// ---------------------------------------------------------------------------
__launch_bounds__(256)
__global__ void ping_head_main(const float* __restrict__ x,
                               const bf16*  __restrict__ wih,   // bf16, 384x1024 row-major
                               const float* __restrict__ bih,
                               const float* __restrict__ bhh,
                               const float* __restrict__ linw,
                               const float* __restrict__ linb,
                               float* __restrict__ out)
{
    __shared__ bf16  lds_x[TILE_M * KCHUNK];    // 8 KB   bf16 x chunk (32x128)
    __shared__ float lds_gi[TILE_M * N_DIM];    // 48 KB  gi = x @ W^T
    __shared__ float lds_red[256];              // 1 KB   reduction scratch

    const int  t    = threadIdx.x;
    const int  lane = t & 31;
    const int  wn   = t >> 5;        // wave = N-group: tiles wn*3 .. wn*3+2
    const long m0   = (long)blockIdx.x * TILE_M;

    const int hl = lane & 15;                 // row (A) / col (B) within tile
    const int kb = (lane < 16) ? 0 : 8;       // K base per ISA 16-bit layout

    // acc[tt*2 + mt]: N-tile tt (0..2) x M-tile mt (0..1)
    v8f acc[6] = {};

    // x staging: thread t owns 16 contiguous floats of row (t>>3) per chunk
    const int xr = t >> 3;            // 0..31
    const int xc = (t & 7) * 16;      // 0,16,...,112
    const float* xrow = x + (m0 + xr) * H_DIM + xc;
    bf16* xsl = &lds_x[xr * KCHUNK + xc];

    // Per-wave B row base pointers (weight rows are gi columns).
    const bf16* wrow[3];
    #pragma unroll
    for (int tt = 0; tt < 3; ++tt)
        wrow[tt] = wih + (long)((wn * 3 + tt) * 16 + hl) * H_DIM + kb;

    for (int kc = 0; kc < H_DIM; kc += KCHUNK) {
        __syncthreads();   // all readers of previous chunk are done
        // Stage 32x128 f32 -> bf16: 4x NT b128 load, 4x b64 store per thread.
        #pragma unroll
        for (int i = 0; i < 4; ++i) {
            v4f xv = __builtin_nontemporal_load((const v4f*)(xrow + kc + i * 4));
            v4bf pk;
            pk.x = (bf16)xv.x; pk.y = (bf16)xv.y;
            pk.z = (bf16)xv.z; pk.w = (bf16)xv.w;
            *(v4bf*)(xsl + i * 4) = pk;
        }
        __syncthreads();   // chunk visible to all waves

        #pragma unroll
        for (int ks = 0; ks < KCHUNK; ks += KSTEP) {
            // Two A fragments: rows 0-15 and rows 16-31 of the M block.
            v16bf a0 = load_frag(&lds_x[hl * KCHUNK + ks + kb]);
            v16bf a1 = load_frag(&lds_x[(16 + hl) * KCHUNK + ks + kb]);

            // Batch all three B fragments first: one clause of 6x b128,
            // distinct register groups, then an uninterrupted WMMA burst.
            v16bf b[3];
            #pragma unroll
            for (int tt = 0; tt < 3; ++tt)
                b[tt] = load_frag(wrow[tt] + kc + ks);

            #pragma unroll
            for (int tt = 0; tt < 3; ++tt) {
                acc[tt * 2 + 0] = __builtin_amdgcn_wmma_f32_16x16x32_bf16(
                    false, a0, false, b[tt], (short)0, acc[tt * 2 + 0], false, false);
                acc[tt * 2 + 1] = __builtin_amdgcn_wmma_f32_16x16x32_bf16(
                    false, a1, false, b[tt], (short)0, acc[tt * 2 + 1], false, false);
            }
        }
    }

    __syncthreads();
    // Spill accumulators to LDS.  C/D layout (ISA 7.12.2):
    // VGPR r, lanes 0-15 -> M=r, N=lane; lanes 16-31 -> M=r+8, N=lane-16.
    #pragma unroll
    for (int tt = 0; tt < 3; ++tt) {
        int ncol = (wn * 3 + tt) * 16 + hl;
        #pragma unroll
        for (int mt = 0; mt < 2; ++mt) {
            #pragma unroll
            for (int r = 0; r < 8; ++r) {
                int m = mt * 16 + r + ((lane & 16) ? 8 : 0);
                lds_gi[m * N_DIM + ncol] = acc[tt * 2 + mt][r];
            }
        }
    }
    __syncthreads();

    // Gate epilogue: 8 threads per row, 16 gate columns each.
    int row = t >> 3;              // 0..31
    int c0  = (t & 7) * 16;        // 0,16,...,112
    float partial = 0.f;
    #pragma unroll 4
    for (int j = 0; j < 16; ++j) {
        int g = c0 + j;
        float ir  = lds_gi[row * N_DIM + g]             + bih[g]           + bhh[g];
        float iz  = lds_gi[row * N_DIM + G_DIM + g]     + bih[G_DIM + g]   + bhh[G_DIM + g];
        float in_ = lds_gi[row * N_DIM + 2 * G_DIM + g] + bih[2 * G_DIM + g];
        float rr = 1.f / (1.f + __expf(-ir));
        float zz = 1.f / (1.f + __expf(-iz));
        float nn = tanhf(in_ + rr * bhh[2 * G_DIM + g]);
        partial += (1.f - zz) * nn * linw[g];
    }
    lds_red[t] = partial;
    __syncthreads();
    if ((t & 7) == 0) {
        float s = 0.f;
        #pragma unroll
        for (int k = 0; k < 8; ++k) s += lds_red[t + k];
        __builtin_nontemporal_store(s + linb[0], out + m0 + row);
    }
}

// ---------------------------------------------------------------------------
extern "C" void kernel_launch(void* const* d_in, const int* in_sizes, int n_in,
                              void* d_out, int out_size, void* d_ws, size_t ws_size,
                              hipStream_t stream) {
    const float* x    = (const float*)d_in[0];
    const float* wih  = (const float*)d_in[1];
    // d_in[2] = weight_hh: unused (hidden state is always zero)
    const float* bih  = (const float*)d_in[3];
    const float* bhh  = (const float*)d_in[4];
    const float* linw = (const float*)d_in[5];
    const float* linb = (const float*)d_in[6];
    float* out = (float*)d_out;

    bf16* wbf = (bf16*)d_ws;                     // 384*1024*2 = 768 KB scratch
    const int nw = 3 * G_DIM * H_DIM;
    wih_to_bf16<<<(nw + 255) / 256, 256, 0, stream>>>(wih, wbf, nw);

    const int M = in_sizes[0] / H_DIM;           // B*L = 65536 rows
    ping_head_main<<<M / TILE_M, 256, 0, stream>>>(x, wbf, bih, bhh, linw, linb, out);
}
